// CrAKNConvV2_69904887710285
// MI455X (gfx1250) — matrix operand order repaired and checked by reading, not compiled, not run on hardware
//
#include <hip/hip_runtime.h>
#include <hip/hip_bf16.h>

// ---------------------------------------------------------------------------
// CDNA5 (gfx1250) fused CrAKN conv. All GEMMs via v_wmma_f32_16x16x32_bf16.
// D = 128, K = 2 hardcoded; N, E taken from in_sizes.
// ---------------------------------------------------------------------------

typedef __attribute__((ext_vector_type(16))) __bf16 v16bf;
typedef __attribute__((ext_vector_type(8)))  float  v8f;

union FragBf { v16bf v; uint4 q[2]; };

__device__ __forceinline__ unsigned short f2bf(float f) {
    unsigned int u = __builtin_bit_cast(unsigned int, f);
    u += 0x7FFFu + ((u >> 16) & 1u);   // round-to-nearest-even
    return (unsigned short)(u >> 16);
}

__device__ __forceinline__ float mishf(float x) {
    float sp = (x > 20.f) ? x : log1pf(__expf(x));
    return x * tanhf(sp);
}

// A fragment: 16x32 bf16 tile from LDS (row-major, lda elements per row).
__device__ __forceinline__ v16bf load_a_frag(const unsigned short* sA, int lda,
                                             int mbase, int kc, int lane) {
    int hi  = lane >> 4;
    int row = mbase + (lane & 15);
    const unsigned short* p = sA + row * lda + kc + hi * 8;
    FragBf f;
    f.q[0] = *(const uint4*)(p);        // K = kc + hi*8 .. +7
    f.q[1] = *(const uint4*)(p + 16);   // K = kc + 16 + hi*8 .. +7
    return f.v;
}

// B fragment: 32x16 bf16 tile from a transposed weight WT[colsOut][kdim].
__device__ __forceinline__ v16bf load_b_frag(const unsigned short* WT, int kdim,
                                             int nbase, int kc, int lane) {
    int hi  = lane >> 4;
    int col = nbase + (lane & 15);
    const unsigned short* p = WT + (size_t)col * kdim + kc + hi * 8;
    FragBf f;
    f.q[0] = *(const uint4*)(p);        // K = kc + hi*8 .. +7
    f.q[1] = *(const uint4*)(p + 16);   // K = kc + 16 + hi*8 .. +7
    return f.v;
}

// 16x16 output tile = A(16 x kdim, LDS bf16) @ WT^T, f32 accumulate.
__device__ __forceinline__ v8f gemm_acc(const unsigned short* sA, int lda,
                                        const unsigned short* WT, int kdim,
                                        int mbase, int nbase, int lane) {
    v8f acc = {};
    for (int kc = 0; kc < kdim; kc += 32) {
        v16bf a = load_a_frag(sA, lda, mbase, kc, lane);
        v16bf b = load_b_frag(WT, kdim, nbase, kc, lane);
        acc = __builtin_amdgcn_wmma_f32_16x16x32_bf16(
            false, a, false, b, (short)0, acc, false, false);
    }
    return acc;
}

// C layout: lane-half hi -> rows mbase+j+hi*8, col = nbase + (lane&15).
__device__ __forceinline__ void store_c(float* sC, int ldc, int mbase, int nbase,
                                        int lane, v8f acc, const float* bias) {
    int hi  = lane >> 4;
    int col = nbase + (lane & 15);
    float b = bias ? bias[col] : 0.f;
#pragma unroll
    for (int j = 0; j < 8; ++j)
        sC[(mbase + j + hi * 8) * ldc + col] = acc[j] + b;
}

__device__ __forceinline__ void store_c_global(float* out, int ldc, int rowbase,
                                               int mbase, int nbase, int lane,
                                               v8f acc, const float* bias, int maxrow) {
    int hi  = lane >> 4;
    int col = nbase + (lane & 15);
    float b = bias ? bias[col] : 0.f;
#pragma unroll
    for (int j = 0; j < 8; ++j) {
        int gr = rowbase + mbase + j + hi * 8;
        if (gr < maxrow) out[(size_t)gr * ldc + col] = acc[j] + b;
    }
}

// LayerNorm + mish over 128 cols, 32 rows in sC -> bf16 in sA. 8 waves x 4 rows.
__device__ __forceinline__ void ln_mish_to_lds(const float* sC, const float* g,
                                               const float* be, unsigned short* sA,
                                               int wave, int lane) {
    int row0 = wave * 4;
#pragma unroll
    for (int i = 0; i < 4; ++i) {
        int row = row0 + i;
        float x[4], s = 0.f, sq = 0.f;
#pragma unroll
        for (int j = 0; j < 4; ++j) {
            x[j] = sC[row * 128 + lane + 32 * j];
            s += x[j]; sq += x[j] * x[j];
        }
#pragma unroll
        for (int o = 16; o; o >>= 1) { s += __shfl_xor(s, o); sq += __shfl_xor(sq, o); }
        float mu = s * 0.0078125f;
        float rs = rsqrtf(sq * 0.0078125f - mu * mu + 1e-5f);
#pragma unroll
        for (int j = 0; j < 4; ++j) {
            int c = lane + 32 * j;
            float y = (x[j] - mu) * rs * g[c] + be[c];
            sA[row * 128 + c] = f2bf(mishf(y));
        }
    }
}

__device__ __forceinline__ void ln_mish_to_global(const float* sC, const float* g,
                                                  const float* be, float* out,
                                                  int n0, int N, int wave, int lane) {
    int row0 = wave * 4;
#pragma unroll
    for (int i = 0; i < 4; ++i) {
        int row = row0 + i;
        float x[4], s = 0.f, sq = 0.f;
#pragma unroll
        for (int j = 0; j < 4; ++j) {
            x[j] = sC[row * 128 + lane + 32 * j];
            s += x[j]; sq += x[j] * x[j];
        }
#pragma unroll
        for (int o = 16; o; o >>= 1) { s += __shfl_xor(s, o); sq += __shfl_xor(sq, o); }
        float mu = s * 0.0078125f;
        float rs = rsqrtf(sq * 0.0078125f - mu * mu + 1e-5f);
        if (n0 + row < N) {
#pragma unroll
            for (int j = 0; j < 4; ++j) {
                int c = lane + 32 * j;
                float y = (x[j] - mu) * rs * g[c] + be[c];
                out[(size_t)(n0 + row) * 128 + c] = mishf(y);
            }
        }
    }
}

// ---------------------------------------------------------------------------
// Prep kernels
// ---------------------------------------------------------------------------
__global__ void transpose_bf16_kernel(const float* __restrict__ src,
                                      unsigned short* __restrict__ dst,
                                      int rows, int cols) {
    size_t mat = (size_t)rows * cols;
    const float* s = src + (size_t)blockIdx.y * mat;
    unsigned short* d = dst + (size_t)blockIdx.y * mat;
    int i = blockIdx.x * 256 + threadIdx.x;
    if (i < rows * cols) {
        int r = i / cols, c = i % cols;
        d[(size_t)c * rows + r] = f2bf(s[i]);   // [out][in] for B-fragments
    }
}

__global__ void zero_kernel(float* __restrict__ p, size_t n) {
    size_t i = (size_t)blockIdx.x * 256 + threadIdx.x;
    if (i < n) p[i] = 0.f;
}

// ---------------------------------------------------------------------------
// Node projections: q = mish(LN(x@Wq+bq)), k likewise, v = x@Wv+bv
// ---------------------------------------------------------------------------
__global__ void __launch_bounds__(256) proj_kernel(
    const float* __restrict__ node,
    const unsigned short* __restrict__ WqT, const float* __restrict__ bq,
    const float* __restrict__ gq, const float* __restrict__ beq,
    const unsigned short* __restrict__ WkT, const float* __restrict__ bk,
    const float* __restrict__ gk, const float* __restrict__ bek,
    const unsigned short* __restrict__ WvT, const float* __restrict__ bv,
    float* __restrict__ qf, float* __restrict__ kf, float* __restrict__ vf, int N)
{
    __shared__ __align__(16) unsigned short sA[32 * 128];
    __shared__ __align__(16) float          sC[32 * 128];
    int n0 = blockIdx.x * 32;
    int lane = threadIdx.x & 31, wave = threadIdx.x >> 5;

    for (int idx = threadIdx.x; idx < 32 * 128; idx += 256) {
        int r = idx >> 7, c = idx & 127;
        int n = n0 + r;
        sA[idx] = f2bf(n < N ? node[(size_t)n * 128 + c] : 0.f);
    }
    __syncthreads();

    // q
    for (int mt = 0; mt < 2; ++mt)
        store_c(sC, 128, mt * 16, wave * 16, lane,
                gemm_acc(sA, 128, WqT, 128, mt * 16, wave * 16, lane), bq);
    __syncthreads();
    ln_mish_to_global(sC, gq, beq, qf, n0, N, wave, lane);
    __syncthreads();

    // k
    for (int mt = 0; mt < 2; ++mt)
        store_c(sC, 128, mt * 16, wave * 16, lane,
                gemm_acc(sA, 128, WkT, 128, mt * 16, wave * 16, lane), bk);
    __syncthreads();
    ln_mish_to_global(sC, gk, bek, kf, n0, N, wave, lane);
    __syncthreads();

    // v (no LN/mish)
    for (int mt = 0; mt < 2; ++mt)
        store_c_global(vf, 128, n0, mt * 16, wave * 16, lane,
                       gemm_acc(sA, 128, WvT, 128, mt * 16, wave * 16, lane), bv, N);
}

// ---------------------------------------------------------------------------
// One (GEMM -> LN -> mish -> GEMM) edge MLP stage, 32 edges per block.
// inA: bf16 LDS A input; outC: f32 LDS output (+b2).
// ---------------------------------------------------------------------------
__device__ __forceinline__ void edge_mlp_stage(
    const unsigned short* inA, float* outC,
    unsigned short* sA, float* sC,
    const unsigned short* W1T, const float* b1,
    const float* g, const float* be,
    const unsigned short* W2T, const float* b2,
    int wave, int lane)
{
    for (int mt = 0; mt < 2; ++mt)
        store_c(sC, 128, mt * 16, wave * 16, lane,
                gemm_acc(inA, 128, W1T, 128, mt * 16, wave * 16, lane), b1);
    __syncthreads();
    ln_mish_to_lds(sC, g, be, sA, wave, lane);
    __syncthreads();
    for (int mt = 0; mt < 2; ++mt)
        store_c(outC, 128, mt * 16, wave * 16, lane,
                gemm_acc(sA, 128, W2T, 128, mt * 16, wave * 16, lane), b2);
    __syncthreads();
}

__global__ void __launch_bounds__(256) edge_kernel(
    const float* __restrict__ qf, const float* __restrict__ kf,
    const float* __restrict__ vf,
    const int* __restrict__ src, const int* __restrict__ dst,
    const float* __restrict__ edge_feat,
    const unsigned short* __restrict__ W1T, const float* __restrict__ b1,
    const float* __restrict__ g, const float* __restrict__ be,
    const unsigned short* __restrict__ W2T, const float* __restrict__ b2,
    float* __restrict__ hbuf, int E, int K)
{
    __shared__ __align__(16) unsigned short sEf[32 * 128];
    __shared__ __align__(16) unsigned short sA [32 * 128];
    __shared__ __align__(16) float          sC  [32 * 128];
    __shared__ __align__(16) float          sPem[32 * 128];

    int e0 = blockIdx.x * 32;
    int kk = blockIdx.y;
    int lane = threadIdx.x & 31, wave = threadIdx.x >> 5;
    const int KD = K * 128;

    // edge features -> bf16 LDS
    for (int idx = threadIdx.x; idx < 32 * 128; idx += 256) {
        int r = idx >> 7, c = idx & 127;
        int e = e0 + r;
        float x = (e < E) ? edge_feat[((size_t)kk * E + e) * 128 + c] : 0.f;
        sEf[idx] = f2bf(x);
    }
    __syncthreads();

    const size_t MS = (size_t)128 * 128;
    // stage 0 (multiplier) -> sPem ; stage 1 (bias) -> sC
    edge_mlp_stage(sEf, sPem, sA, sC,
                   W1T + (size_t)(0 * K + kk) * MS, b1 + (0 * K + kk) * 128,
                   g   + (0 * K + kk) * 128,        be + (0 * K + kk) * 128,
                   W2T + (size_t)(0 * K + kk) * MS, b2 + (0 * K + kk) * 128,
                   wave, lane);
    edge_mlp_stage(sEf, sC, sA, sC,
                   W1T + (size_t)(1 * K + kk) * MS, b1 + (1 * K + kk) * 128,
                   g   + (1 * K + kk) * 128,        be + (1 * K + kk) * 128,
                   W2T + (size_t)(1 * K + kk) * MS, b2 + (1 * K + kk) * 128,
                   wave, lane);

    // r = (q[src]-k[dst])/sqrt(||.||^2 + 1e-8) * pem + peb  -> bf16 sA
    {
        int row0 = wave * 4;
#pragma unroll
        for (int i = 0; i < 4; ++i) {
            int row = row0 + i;
            int e = e0 + row;
            int se = (e < E) ? src[e] : 0;
            int de = (e < E) ? dst[e] : 0;
            float rel[4], sq = 0.f;
#pragma unroll
            for (int j = 0; j < 4; ++j) {
                int c = lane + 32 * j;
                rel[j] = qf[(size_t)se * 128 + c] - kf[(size_t)de * 128 + c];
                sq += rel[j] * rel[j];
            }
#pragma unroll
            for (int o = 16; o; o >>= 1) sq += __shfl_xor(sq, o);
            float rs = 1.f / sqrtf(sq + 1e-8f);
#pragma unroll
            for (int j = 0; j < 4; ++j) {
                int c = lane + 32 * j;
                float r = rel[j] * rs * sPem[row * 128 + c] + sC[row * 128 + c];
                sA[row * 128 + c] = f2bf(r);
            }
        }
    }
    __syncthreads();

    // stage 2 (weight encoding): w -> sC
    edge_mlp_stage(sA, sC, sA, sC,
                   W1T + (size_t)(2 * K + kk) * MS, b1 + (2 * K + kk) * 128,
                   g   + (2 * K + kk) * 128,        be + (2 * K + kk) * 128,
                   W2T + (size_t)(2 * K + kk) * MS, b2 + (2 * K + kk) * 128,
                   wave, lane);

    // val = w * v[dst]; segment-sum via f32 global atomics into concat layout
    {
        int row0 = wave * 4;
#pragma unroll
        for (int i = 0; i < 4; ++i) {
            int row = row0 + i;
            int e = e0 + row;
            if (e < E) {
                int d = dst[e];
#pragma unroll
                for (int j = 0; j < 4; ++j) {
                    int c = lane + 32 * j;
                    float val = sC[row * 128 + c] * vf[(size_t)d * 128 + c];
                    atomicAdd(&hbuf[(size_t)d * KD + kk * 128 + c], val);
                }
            }
        }
    }
}

// ---------------------------------------------------------------------------
// Combiner: out = mish(h @ comb_W1 + comb_b1) @ comb_W2   (h: [N,256])
// ---------------------------------------------------------------------------
__global__ void __launch_bounds__(256) comb_kernel(
    const float* __restrict__ hbuf,
    const unsigned short* __restrict__ W1T, const float* __restrict__ b1,
    const unsigned short* __restrict__ W2T,
    float* __restrict__ out, int N)
{
    __shared__ __align__(16) unsigned short sH[32 * 256];
    __shared__ __align__(16) float          sC[32 * 256];
    int n0 = blockIdx.x * 32;
    int lane = threadIdx.x & 31, wave = threadIdx.x >> 5;

    for (int idx = threadIdx.x; idx < 32 * 256; idx += 256) {
        int r = idx >> 8, c = idx & 255;
        int n = n0 + r;
        sH[idx] = f2bf(n < N ? hbuf[(size_t)n * 256 + c] : 0.f);
    }
    __syncthreads();

    for (int mt = 0; mt < 2; ++mt)
        for (int nt = 0; nt < 2; ++nt) {
            int nbase = (wave + nt * 8) * 16;
            store_c(sC, 256, mt * 16, nbase, lane,
                    gemm_acc(sH, 256, W1T, 256, mt * 16, nbase, lane), b1);
        }
    __syncthreads();

    // mish (no LN) -> bf16
    {
        int row0 = wave * 4;
#pragma unroll
        for (int i = 0; i < 4; ++i) {
            int row = row0 + i;
#pragma unroll
            for (int j = 0; j < 8; ++j) {
                int c = lane + 32 * j;
                sH[row * 256 + c] = f2bf(mishf(sC[row * 256 + c]));
            }
        }
    }
    __syncthreads();

    for (int mt = 0; mt < 2; ++mt)
        store_c_global(out, 128, n0, mt * 16, wave * 16, lane,
                       gemm_acc(sH, 256, W2T, 256, mt * 16, wave * 16, lane),
                       nullptr, N);
}

// ---------------------------------------------------------------------------
// Host-side launch
// ---------------------------------------------------------------------------
extern "C" void kernel_launch(void* const* d_in, const int* in_sizes, int n_in,
                              void* d_out, int out_size, void* d_ws, size_t ws_size,
                              hipStream_t stream) {
    const float* node      = (const float*)d_in[0];
    const int*   src       = (const int*)d_in[1];
    const int*   dst       = (const int*)d_in[2];
    const float* edge_feat = (const float*)d_in[3];
    const float* Wq = (const float*)d_in[4],  *bq = (const float*)d_in[5];
    const float* gq = (const float*)d_in[6],  *beq = (const float*)d_in[7];
    const float* Wk = (const float*)d_in[8],  *bk = (const float*)d_in[9];
    const float* gk = (const float*)d_in[10], *bek = (const float*)d_in[11];
    const float* Wv = (const float*)d_in[12], *bv = (const float*)d_in[13];
    const float* mlp_W1 = (const float*)d_in[14], *mlp_b1 = (const float*)d_in[15];
    const float* mlp_g  = (const float*)d_in[16], *mlp_be = (const float*)d_in[17];
    const float* mlp_W2 = (const float*)d_in[18], *mlp_b2 = (const float*)d_in[19];
    const float* comb_W1 = (const float*)d_in[20], *comb_b1 = (const float*)d_in[21];
    const float* comb_W2 = (const float*)d_in[22];

    const int D  = 128;
    const int N  = in_sizes[0] / D;
    const int E  = in_sizes[1];
    const int K  = in_sizes[3] / (E * D);   // = 2
    const int KD = K * D;                   // = 256

    // workspace carve-out (256B aligned)
    char*  ws  = (char*)d_ws;
    size_t off = 0;
    auto alloc = [&](size_t bytes) -> void* {
        void* p = ws + off;
        off += (bytes + 255) & ~(size_t)255;
        return p;
    };
    float* qf   = (float*)alloc((size_t)N * D * 4);
    float* kf   = (float*)alloc((size_t)N * D * 4);
    float* vf   = (float*)alloc((size_t)N * D * 4);
    float* hbuf = (float*)alloc((size_t)N * KD * 4);
    unsigned short* WqT     = (unsigned short*)alloc((size_t)D * D * 2);
    unsigned short* WkT     = (unsigned short*)alloc((size_t)D * D * 2);
    unsigned short* WvT     = (unsigned short*)alloc((size_t)D * D * 2);
    unsigned short* W1T     = (unsigned short*)alloc((size_t)3 * K * D * D * 2);
    unsigned short* W2T     = (unsigned short*)alloc((size_t)3 * K * D * D * 2);
    unsigned short* combW1T = (unsigned short*)alloc((size_t)KD * KD * 2);
    unsigned short* combW2T = (unsigned short*)alloc((size_t)KD * D * 2);

    dim3 blk(256);
    int mat128 = (D * D + 255) / 256;
    transpose_bf16_kernel<<<dim3(mat128, 1), blk, 0, stream>>>(Wq, WqT, D, D);
    transpose_bf16_kernel<<<dim3(mat128, 1), blk, 0, stream>>>(Wk, WkT, D, D);
    transpose_bf16_kernel<<<dim3(mat128, 1), blk, 0, stream>>>(Wv, WvT, D, D);
    transpose_bf16_kernel<<<dim3(mat128, 3 * K), blk, 0, stream>>>(mlp_W1, W1T, D, D);
    transpose_bf16_kernel<<<dim3(mat128, 3 * K), blk, 0, stream>>>(mlp_W2, W2T, D, D);
    transpose_bf16_kernel<<<dim3((KD * KD + 255) / 256, 1), blk, 0, stream>>>(comb_W1, combW1T, KD, KD);
    transpose_bf16_kernel<<<dim3((KD * D + 255) / 256, 1), blk, 0, stream>>>(comb_W2, combW2T, KD, D);

    size_t hN = (size_t)N * KD;
    zero_kernel<<<dim3((unsigned)((hN + 255) / 256)), blk, 0, stream>>>(hbuf, hN);

    proj_kernel<<<dim3((N + 31) / 32), blk, 0, stream>>>(
        node, WqT, bq, gq, beq, WkT, bk, gk, bek, WvT, bv, qf, kf, vf, N);

    edge_kernel<<<dim3((E + 31) / 32, K), blk, 0, stream>>>(
        qf, kf, vf, src, dst, edge_feat,
        W1T, mlp_b1, mlp_g, mlp_be, W2T, mlp_b2, hbuf, E, K);

    comb_kernel<<<dim3((N + 31) / 32), blk, 0, stream>>>(
        hbuf, combW1T, comb_b1, combW2T, (float*)d_out, N);
}
